// GRUMemoryUpdate_86225763435163
// MI455X (gfx1250) — compile-verified, matrix operand order checked by reading, not run on hardware
//
#include <hip/hip_runtime.h>
#include <hip/hip_bf16.h>
#include <stdint.h>

typedef __attribute__((ext_vector_type(16))) _Float16 v16h;
typedef __attribute__((ext_vector_type(8)))  float    v8f;

#define B_ 8
#define T_ 12
#define N_ 4096
#define F_ 64
#define H_ 64
#define G_ 192   // 3*H

#define LOG2E_ 1.4426950408889634f

// K-pair base for 16-bit A/B fragment element pair v (ISA 7.12.2, 16-bit A 16x32)
__device__ __forceinline__ int kpair(int v, int lane) {
  return ((v < 4) ? (2 * v) : (16 + 2 * (v - 4))) + ((lane & 16) ? 8 : 0);
}

// Build A fragment (16x32 f16) for K-chunk c from a row-major f32 [16][64] LDS tile
__device__ __forceinline__ v16h afrag_f32(const float* src, int lane, int c) {
  v16h a; int m = lane & 15;
  #pragma unroll
  for (int v = 0; v < 8; ++v) {
    int k = 32 * c + kpair(v, lane);
    a[2 * v]     = (_Float16)src[m * 64 + k];
    a[2 * v + 1] = (_Float16)src[m * 64 + k + 1];
  }
  return a;
}

// Same, but from a row-major f16 [16][64] LDS tile (the hidden state)
__device__ __forceinline__ v16h afrag_f16(const _Float16* src, int lane, int c) {
  v16h a; int m = lane & 15;
  #pragma unroll
  for (int v = 0; v < 8; ++v) {
    int k = 32 * c + kpair(v, lane);
    a[2 * v]     = src[m * 64 + k];
    a[2 * v + 1] = src[m * 64 + k + 1];
  }
  return a;
}

// Fast transcendentals on the recurrence critical path: v_exp_f32 + v_rcp_f32,
// no IEEE divide expansion.
__device__ __forceinline__ float fast_sigmoid(float x) {
  float e = __builtin_amdgcn_exp2f(-LOG2E_ * x);
  return __builtin_amdgcn_rcpf(1.0f + e);
}
__device__ __forceinline__ float fast_tanh(float x) {
  // tanh(x) = 1 - 2 / (exp(2x) + 1)
  float e = __builtin_amdgcn_exp2f((2.0f * LOG2E_) * x);
  return __builtin_fmaf(-2.0f, __builtin_amdgcn_rcpf(1.0f + e), 1.0f);
}

__global__ __launch_bounds__(384)
void gru_mem_kernel(const float* __restrict__ memory,
                    const float* __restrict__ data,
                    const int*   __restrict__ indices,
                    const float* __restrict__ W_ih,
                    const float* __restrict__ W_hh,
                    const float* __restrict__ b_ih,
                    const float* __restrict__ b_hh,
                    float* __restrict__ out)
{
  __shared__ float    xbuf[2][16][F_];   // async-load ring for x[n] (rows 12..15 = 0)
  __shared__ float    gis[16][G_];       // input-side gates  (f32 from WMMA C)
  __shared__ float    ghs[16][G_];       // hidden-side gates
  __shared__ _Float16 hh[16][H_];        // h in f16 (A operand), rows 12..15 = 0
  __shared__ float    hf[T_][H_];        // h master copy in f32

  const int tid  = threadIdx.x;
  const int lane = tid & 31;
  const int wv   = tid >> 5;             // wave id: gate tile 0..11, also t-row fetched
  const int b    = blockIdx.x;

  // ---- init hidden state + zero padding rows ----
  const int idx = indices[b];
  for (int e = tid; e < T_ * H_; e += 384) {
    int t = e >> 6, j = e & 63;
    float h0 = memory[idx * H_ + j];
    hf[t][j] = h0;
    hh[t][j] = (_Float16)h0;
  }
  for (int e = tid; e < 4 * H_; e += 384) {
    int t = 12 + (e >> 6), j = e & 63;
    hh[t][j] = (_Float16)0.0f;
    xbuf[0][t][j] = 0.0f;
    xbuf[1][t][j] = 0.0f;
  }

  // ---- per-wave weight B-fragments, pinned in registers for the whole loop ----
  // gi[t,g] = sum_f x[t,f] * W_ih[g,f]  =>  B[k=f][n=g] = W_ih[g][f]
  v16h bih[2], bhh[2];
  {
    int g = wv * 16 + (lane & 15);       // output (gate) column of this lane
    #pragma unroll
    for (int c = 0; c < 2; ++c) {
      #pragma unroll
      for (int v = 0; v < 8; ++v) {
        int k = 32 * c + kpair(v, lane);
        bih[c][2 * v]     = (_Float16)W_ih[g * F_ + k];
        bih[c][2 * v + 1] = (_Float16)W_ih[g * F_ + k + 1];
        bhh[c][2 * v]     = (_Float16)W_hh[g * H_ + k];
        bhh[c][2 * v + 1] = (_Float16)W_hh[g * H_ + k + 1];
      }
    }
  }
  const float biasI = b_ih[wv * 16 + (lane & 15)];
  const float biasH = b_hh[wv * 16 + (lane & 15)];

  // byte offset of data[b, t=wv, n=0, f=0]
  const uint64_t gbase   = (uint64_t)(uintptr_t)data;
  const uint32_t rowoff0 = (uint32_t)((((size_t)b * T_ + wv) * N_) * F_ * 4u);

  __syncthreads();

  // prologue: async prefetch x[0] into ring slot 0 (each wave: its row, 8B/lane)
  {
    uint32_t ldsa = (uint32_t)(uintptr_t)&xbuf[0][wv][lane * 2];
    uint32_t goff = rowoff0 + (uint32_t)lane * 8u;
    asm volatile("global_load_async_to_lds_b64 %0, %1, %2"
                 :: "v"(ldsa), "v"(goff), "s"(gbase) : "memory");
  }

  for (int n = 0; n < N_; ++n) {
    const int cur = n & 1;
    if (n + 1 < N_) {
      uint32_t ldsa = (uint32_t)(uintptr_t)&xbuf[cur ^ 1][wv][lane * 2];
      uint32_t goff = rowoff0 + (uint32_t)(n + 1) * (F_ * 4u) + (uint32_t)lane * 8u;
      asm volatile("global_load_async_to_lds_b64 %0, %1, %2"
                   :: "v"(ldsa), "v"(goff), "s"(gbase) : "memory");
      asm volatile("s_wait_asynccnt 0x1" ::: "memory");   // x[n] landed; x[n+1] in flight
    } else {
      asm volatile("s_wait_asynccnt 0x0" ::: "memory");
    }
    __syncthreads();   // x[n] visible to all waves; hh updated by previous combine

    // A fragments (shared by all 12 waves; B differs per wave)
    v16h xa0 = afrag_f32(&xbuf[cur][0][0], lane, 0);
    v16h xa1 = afrag_f32(&xbuf[cur][0][0], lane, 1);
    v16h ha0 = afrag_f16(&hh[0][0], lane, 0);
    v16h ha1 = afrag_f16(&hh[0][0], lane, 1);

    v8f accI, accH;
    #pragma unroll
    for (int v = 0; v < 8; ++v) { accI[v] = biasI; accH[v] = biasH; }

    accI = __builtin_amdgcn_wmma_f32_16x16x32_f16(false, xa0, false, bih[0], (short)0, accI, false, false);
    accI = __builtin_amdgcn_wmma_f32_16x16x32_f16(false, xa1, false, bih[1], (short)0, accI, false, false);
    accH = __builtin_amdgcn_wmma_f32_16x16x32_f16(false, ha0, false, bhh[0], (short)0, accH, false, false);
    accH = __builtin_amdgcn_wmma_f32_16x16x32_f16(false, ha1, false, bhh[1], (short)0, accH, false, false);

    // scatter C tiles to LDS: row = v + 8*(lane>=16), col = wv*16 + lane%16
    {
      int col = wv * 16 + (lane & 15);
      int r0  = (lane & 16) ? 8 : 0;
      #pragma unroll
      for (int v = 0; v < 8; ++v) {
        gis[r0 + v][col] = accI[v];
        ghs[r0 + v][col] = accH[v];
      }
    }
    __syncthreads();   // all gate tiles ready

    // gate combine + hidden-state update: 12*64 = 768 elements, 2 per thread
    #pragma unroll
    for (int q = 0; q < 2; ++q) {
      int e = tid + q * 384;
      int t = e >> 6, j = e & 63;
      float r  = fast_sigmoid(gis[t][j]      + ghs[t][j]);
      float z  = fast_sigmoid(gis[t][H_ + j] + ghs[t][H_ + j]);
      float nv = fast_tanh(__builtin_fmaf(r, ghs[t][2 * H_ + j], gis[t][2 * H_ + j]));
      float hp = hf[t][j];
      float hn = __builtin_fmaf(z, hp - nv, nv);  // (1-z)*nv + z*hp
      hf[t][j] = hn;
      hh[t][j] = (_Float16)hn;
    }
    // next iteration's top barrier orders hh writes before the A-fragment reads
  }

  __syncthreads();
  if (tid < H_) {
    float s = 0.0f;
    #pragma unroll
    for (int t = 0; t < T_; ++t) s += hf[t][tid];
    out[b * H_ + tid] = s * (1.0f / 12.0f);
  }
}

extern "C" void kernel_launch(void* const* d_in, const int* in_sizes, int n_in,
                              void* d_out, int out_size, void* d_ws, size_t ws_size,
                              hipStream_t stream) {
  (void)in_sizes; (void)n_in; (void)out_size; (void)d_ws; (void)ws_size;
  const float* memory  = (const float*)d_in[0];
  const float* data    = (const float*)d_in[1];
  const int*   indices = (const int*)d_in[2];
  const float* W_ih    = (const float*)d_in[3];
  const float* W_hh    = (const float*)d_in[4];
  const float* b_ih    = (const float*)d_in[5];
  const float* b_hh    = (const float*)d_in[6];
  // d_in[7] = update_counter (unused by the reference math)
  float* out = (float*)d_out;

  gru_mem_kernel<<<dim3(B_), dim3(384), 0, stream>>>(
      memory, data, indices, W_ih, W_hh, b_ih, b_hh, out);
}
